// GCNN_2147483648236
// MI455X (gfx1250) — compile-verified, matrix operand order checked by reading, not run on hardware
//
#include <hip/hip_runtime.h>
#include <hip/hip_bf16.h>
#include <math.h>

// ---------------------------------------------------------------------------
// GCNN for MI455X (gfx1250): flash-style on-the-fly edge-kernel + bf16 WMMA.
// B fragments built with ds_load_b128 / global_load_b128 (K-contiguous
// layouts) instead of scalar u16 gathers.
// ---------------------------------------------------------------------------

typedef __attribute__((ext_vector_type(16))) __bf16          v16bf;
typedef __attribute__((ext_vector_type(8)))  float           v8f;
typedef __attribute__((ext_vector_type(16))) unsigned short  u16x16;
typedef __attribute__((ext_vector_type(4)))  unsigned int    v4u;
typedef __attribute__((ext_vector_type(8)))  unsigned int    v8u;

#define B_   8
#define N_   2048
#define F0_  29
#define NDIL 4
#define KPAD 40   // padded u16 row stride of transposed LDS tile (80B, 16B-aligned)

__device__ __forceinline__ unsigned short f2bf(float x) {
  union { float f; unsigned u; } v; v.f = x;
  unsigned r = v.u + 0x7FFFu + ((v.u >> 16) & 1u);   // round-to-nearest-even
  return (unsigned short)(r >> 16);
}
__device__ __forceinline__ float bf2f(unsigned short h) {
  union { unsigned u; float f; } v; v.u = ((unsigned)h) << 16;
  return v.f;
}
// K index for element e of a 16-bit 16x32 A fragment (ISA 7.12.2).
__device__ __forceinline__ int kmap(int e, int half) {
  int p = e >> 1, o = e & 1;
  return ((p & 3) * 2 + o) + ((p >> 2) * 16) + half * 8;
}
__device__ __forceinline__ v16bf frag_from(v4u lo, v4u hi) {
  v8u w = {lo[0], lo[1], lo[2], lo[3], hi[0], hi[1], hi[2], hi[3]};
  return __builtin_bit_cast(v16bf, w);
}
__device__ __forceinline__ float tanh_fast(float x) {
  // tanh(x) = 1 - 2/(exp(2x)+1)   (v_exp_f32 + v_rcp_f32, branch-free)
  float e = __expf(2.0f * x);
  return 1.0f - 2.0f * __builtin_amdgcn_rcpf(e + 1.0f);
}

// --------------------------- small prep kernels ----------------------------

__global__ void k_sigma(const float* __restrict__ V, const float* __restrict__ we_w,
                        const float* __restrict__ we_b, float* __restrict__ sigma) {
  int i = blockIdx.x * blockDim.x + threadIdx.x;
  if (i >= B_ * N_) return;
  float x = we_b[0];
  #pragma unroll
  for (int f = 0; f < F0_; ++f) x += V[i * F0_ + f] * we_w[f];
  sigma[i] = (x > 20.f) ? x : log1pf(__expf(x));   // softplus
}

__global__ void k_convV(const float* __restrict__ V, unsigned short* __restrict__ Hb) {
  int idx = blockIdx.x * blockDim.x + threadIdx.x;
  if (idx >= B_ * N_ * 32) return;
  int i = idx >> 5, f = idx & 31;
  Hb[idx] = f2bf((f < F0_) ? V[i * F0_ + f] : 0.f);
}

// W [d][Fin][Fout] fp32  ->  WbT [fout][d*Fpad+f] bf16 (K contiguous)
__global__ void k_convW(const float* __restrict__ W, unsigned short* __restrict__ WbT,
                        int Fin, int Fpad, int Fout) {
  int idx = blockIdx.x * blockDim.x + threadIdx.x;
  int Kp = NDIL * Fpad;
  if (idx >= Kp * Fout) return;
  int fo = idx / Kp, k = idx - fo * Kp;
  int d = k / Fpad, f = k - d * Fpad;
  WbT[idx] = f2bf((f < Fin) ? W[(d * Fin + f) * Fout + fo] : 0.f);
}

__global__ void k_zero(float* __restrict__ p, int n) {
  int i = blockIdx.x * blockDim.x + threadIdx.x;
  if (i < n) p[i] = 0.f;
}

// --------------- kernel A: AH[d,b,i,f] = sum_j exp-kernel * H -------------
// One wave per (i-tile, batch, dilation). Ad tile generated in registers,
// H tile staged K-transposed in LDS so each B fragment is 2x ds_load_b128.

template <int NT>   // NT = Fpad/16
__global__ __launch_bounds__(32) void k_graphAH(
    const float* __restrict__ C, const float* __restrict__ sigma,
    const unsigned short* __restrict__ Hin, unsigned short* __restrict__ AHb) {
  constexpr int FP = NT * 16;
  const int it   = blockIdx.x;
  const int b    = blockIdx.y;
  const int d    = blockIdx.z;
  const int lane = threadIdx.x;
  const int half = lane >> 4;
  const int m    = lane & 15;
  const int i0   = it * 16;

  __shared__ float sC[32 * 3];
  __shared__ __align__(16) unsigned short sHT[FP * KPAD];  // [f][k] transposed

  const int   rowi = b * N_ + i0 + m;
  const float cix = C[rowi * 3 + 0], ciy = C[rowi * 3 + 1], ciz = C[rowi * 3 + 2];
  const float inv = 1.0f / (sigma[rowi] * (float)(1 << d));   // dil = 1,2,4,8

  v8f acc[NT] = {};

  const unsigned int* H32 = (const unsigned int*)(Hin + (size_t)(b * N_) * FP);
  for (int j0 = 0; j0 < N_; j0 += 32) {
    __syncthreads();
    // stage coords of 32 j-nodes
    int j = j0 + lane;
    sC[lane * 3 + 0] = C[(b * N_ + j) * 3 + 0];
    sC[lane * 3 + 1] = C[(b * N_ + j) * 3 + 1];
    sC[lane * 3 + 2] = C[(b * N_ + j) * 3 + 2];
    // stage H tile transposed: 32 rows x FP bf16 -> sHT[f][j]
    const unsigned int* src = H32 + (size_t)j0 * FP / 2;
    if (j0 + 32 < N_)   // gfx1250 global_prefetch of next tile
      __builtin_prefetch((const char*)(src + 16 * FP) + lane * 64, 0, 1);
    #pragma unroll
    for (int t = 0; t < FP / 2; ++t) {
      int idx = t * 32 + lane;                 // dword = j*(FP/2) + fpair
      unsigned int u = src[idx];
      int jr = idx / (FP / 2);
      int fp = idx & (FP / 2 - 1);
      sHT[(2 * fp)     * KPAD + jr] = (unsigned short)(u & 0xffffu);
      sHT[(2 * fp + 1) * KPAD + jr] = (unsigned short)(u >> 16);
    }
    __syncthreads();

    // A fragment: Ad(i0+m, j0+k) generated on the fly
    u16x16 ua;
    #pragma unroll
    for (int e = 0; e < 16; ++e) {
      int k = kmap(e, half);
      float dx = cix - sC[k * 3 + 0];
      float dy = ciy - sC[k * 3 + 1];
      float dz = ciz - sC[k * 3 + 2];
      float dist = __builtin_amdgcn_sqrtf(dx * dx + dy * dy + dz * dz + 1e-8f);
      ua[e] = f2bf(__expf(-dist * inv));
    }
    v16bf afrag = __builtin_bit_cast(v16bf, ua);

    #pragma unroll
    for (int t = 0; t < NT; ++t) {
      // B[k][n] = H[j0+k][t*16+n]: elements 0..7 -> k=half*8+0..7,
      // 8..15 -> k=16+half*8+0..7  => two ds_load_b128 from sHT
      const unsigned short* rb = &sHT[(t * 16 + m) * KPAD + half * 8];
      v4u lo = *(const v4u*)(rb);
      v4u hi = *(const v4u*)(rb + 16);
      v16bf bfrag = frag_from(lo, hi);
      acc[t] = __builtin_amdgcn_wmma_f32_16x16x32_bf16(
          false, afrag, false, bfrag, (short)0, acc[t], false, false);
    }
  }

  // write AH as bf16, K-major layout [(b*N+i)][d*FP + f] for the next matmul
  const int K = NDIL * FP;
  #pragma unroll
  for (int t = 0; t < NT; ++t) {
    #pragma unroll
    for (int r = 0; r < 8; ++r) {
      int i = i0 + r + 8 * half;           // C/D layout: row = r + 8*(lane>=16)
      int f = t * 16 + m;                  // col = lane & 15
      AHb[(size_t)(b * N_ + i) * K + d * FP + f] = f2bf(acc[t][r]);
    }
  }
}

// ---- kernel B: H' = BN(tanh(AH @ W / 4 + mean_d(b)))  (WMMA over K) ------
// Both operands are K-contiguous => all fragments are 2x global_load_b128.

__global__ __launch_bounds__(32) void k_convOut(
    const unsigned short* __restrict__ AHb, const unsigned short* __restrict__ WbT,
    const float* __restrict__ bl, const float* __restrict__ gamma,
    const float* __restrict__ beta, const float* __restrict__ mean,
    const float* __restrict__ var, unsigned short* __restrict__ Hout,
    int K, int Fout) {
  const int tile = blockIdx.x;             // (B*N)/16 row tiles
  const int fo0  = blockIdx.y * 16;
  const int lane = threadIdx.x;
  const int half = lane >> 4;
  const int n    = lane & 15;              // A-row == B/D-col for this lane

  v8f acc = {};
  const unsigned short* arow = AHb + (size_t)(tile * 16 + n) * K;
  const unsigned short* brow = WbT + (size_t)(fo0 + n) * K;
  for (int k0 = 0; k0 < K; k0 += 32) {
    const unsigned short* pa = arow + k0 + half * 8;
    v16bf afrag = frag_from(*(const v4u*)pa, *(const v4u*)(pa + 16));
    const unsigned short* pb = brow + k0 + half * 8;
    v16bf bfrag = frag_from(*(const v4u*)pb, *(const v4u*)(pb + 16));
    acc = __builtin_amdgcn_wmma_f32_16x16x32_bf16(
        false, afrag, false, bfrag, (short)0, acc, false, false);
  }

  int f = fo0 + n;
  float bbar = 0.25f * (bl[f] + bl[Fout + f] + bl[2 * Fout + f] + bl[3 * Fout + f]);
  float sc = gamma[f] * __builtin_amdgcn_rsqf(var[f] + 1e-3f);
  float sh = beta[f] - mean[f] * sc;
  #pragma unroll
  for (int r = 0; r < 8; ++r) {
    int row_o = tile * 16 + r + 8 * half;
    float h = tanh_fast(acc[r] * 0.25f + bbar);
    Hout[(size_t)row_o * Fout + f] = f2bf(h * sc + sh);
  }
}

// ------------------------------- tail ops ----------------------------------

__global__ void k_pool(const unsigned short* __restrict__ H, float* __restrict__ P) {
  const int F = 256;
  int idx = blockIdx.x * blockDim.x + threadIdx.x;
  if (idx >= B_ * (N_ / 4) * F) return;
  int f  = idx % F;
  int n4 = (idx / F) % (N_ / 4);
  int b  = idx / (F * (N_ / 4));
  float mx = -3.4e38f;
  #pragma unroll
  for (int q = 0; q < 4; ++q)
    mx = fmaxf(mx, bf2f(H[(size_t)(b * N_ + n4 * 4 + q) * F + f]));
  P[idx] = mx;                              // == flatten(B, (N/4)*F)
}

__global__ __launch_bounds__(512) void k_fc1(const float* __restrict__ P,
                                             const float* __restrict__ W,
                                             float* __restrict__ accOut) {
  const int DIN = (N_ / 4) * 256;           // 131072
  const int fo = threadIdx.x;               // 0..511
  const int kbase = blockIdx.x * 2048;      // 64 blocks * 2048 k
  __shared__ float pT[8 * 256];
  float acc[8] = {};
  for (int sub = 0; sub < 2048; sub += 256) {
    __syncthreads();
    for (int idx = threadIdx.x; idx < 8 * 256; idx += 512) {
      int b = idx >> 8, kk = idx & 255;
      pT[idx] = P[(size_t)b * DIN + kbase + sub + kk];
    }
    __syncthreads();
    for (int kk = 0; kk < 256; ++kk) {
      float w = W[(size_t)(kbase + sub + kk) * 512 + fo];   // coalesced over fo
      #pragma unroll
      for (int b = 0; b < 8; ++b) acc[b] += pT[b * 256 + kk] * w;
    }
  }
  #pragma unroll
  for (int b = 0; b < 8; ++b) atomicAdd(&accOut[b * 512 + fo], acc[b]);
}

__global__ __launch_bounds__(256) void k_fc2(const float* __restrict__ acc1,
                                             const float* __restrict__ b1,
                                             const float* __restrict__ W2,
                                             const float* __restrict__ b2,
                                             float* __restrict__ out) {
  __shared__ float F1[8 * 512];
  for (int idx = threadIdx.x; idx < 8 * 512; idx += 256) {
    int k = idx & 511;
    F1[idx] = fmaxf(0.f, acc1[idx] + b1[k]);
  }
  __syncthreads();
  int fo = threadIdx.x;
  for (int b = 0; b < 8; ++b) {
    float s = b2[fo];
    for (int k = 0; k < 512; ++k) s += F1[b * 512 + k] * W2[(size_t)k * 256 + fo];
    out[b * 256 + fo] = fmaxf(0.f, s);
  }
}

// ------------------------------- launcher ----------------------------------

extern "C" void kernel_launch(void* const* d_in, const int* in_sizes, int n_in,
                              void* d_out, int out_size, void* d_ws, size_t ws_size,
                              hipStream_t stream) {
  (void)in_sizes; (void)n_in; (void)out_size; (void)ws_size;
  const float* V    = (const float*)d_in[0];
  const float* C    = (const float*)d_in[1];
  const float* we_w = (const float*)d_in[2];
  const float* we_b = (const float*)d_in[3];
  const float* Wl[3]  = {(const float*)d_in[4],  (const float*)d_in[10], (const float*)d_in[16]};
  const float* bll[3] = {(const float*)d_in[5],  (const float*)d_in[11], (const float*)d_in[17]};
  const float* gl[3]  = {(const float*)d_in[6],  (const float*)d_in[12], (const float*)d_in[18]};
  const float* bel[3] = {(const float*)d_in[7],  (const float*)d_in[13], (const float*)d_in[19]};
  const float* ml[3]  = {(const float*)d_in[8],  (const float*)d_in[14], (const float*)d_in[20]};
  const float* vl[3]  = {(const float*)d_in[9],  (const float*)d_in[15], (const float*)d_in[21]};
  const float* fcW1 = (const float*)d_in[22];
  const float* fcb1 = (const float*)d_in[23];
  const float* fcW2 = (const float*)d_in[24];
  const float* fcb2 = (const float*)d_in[25];

  char* ws = (char*)d_ws;
  size_t off = 0;
  float* sigma          = (float*)(ws + off);          off += 65536;       // 8*2048 f32
  unsigned short* HbA   = (unsigned short*)(ws + off); off += 8388608;     // bf16 H ping
  unsigned short* HbB   = (unsigned short*)(ws + off); off += 8388608;     // bf16 H pong
  unsigned short* AHb   = (unsigned short*)(ws + off); off += 16777216;    // bf16 AH
  unsigned short* WbT   = (unsigned short*)(ws + off); off += 262144;      // bf16 W^T
  float* P              = (float*)(ws + off);          off += 4194304;     // pooled
  float* FC1            = (float*)(ws + off);          off += 16384;       // fc1 acc

  k_sigma<<<(B_ * N_ + 255) / 256, 256, 0, stream>>>(V, we_w, we_b, sigma);
  k_convV<<<(B_ * N_ * 32 + 255) / 256, 256, 0, stream>>>(V, HbA);

  dim3 gA(N_ / 16, B_, NDIL);

  // layer 1: Fin 29 (pad 32), Fout 64
  k_convW<<<(NDIL * 32 * 64 + 255) / 256, 256, 0, stream>>>(Wl[0], WbT, 29, 32, 64);
  k_graphAH<2><<<gA, 32, 0, stream>>>(C, sigma, HbA, AHb);
  k_convOut<<<dim3(B_ * N_ / 16, 64 / 16), 32, 0, stream>>>(
      AHb, WbT, bll[0], gl[0], bel[0], ml[0], vl[0], HbB, NDIL * 32, 64);

  // layer 2: Fin 64, Fout 128
  k_convW<<<(NDIL * 64 * 128 + 255) / 256, 256, 0, stream>>>(Wl[1], WbT, 64, 64, 128);
  k_graphAH<4><<<gA, 32, 0, stream>>>(C, sigma, HbB, AHb);
  k_convOut<<<dim3(B_ * N_ / 16, 128 / 16), 32, 0, stream>>>(
      AHb, WbT, bll[1], gl[1], bel[1], ml[1], vl[1], HbA, NDIL * 64, 128);

  // layer 3: Fin 128, Fout 256
  k_convW<<<(NDIL * 128 * 256 + 255) / 256, 256, 0, stream>>>(Wl[2], WbT, 128, 128, 256);
  k_graphAH<8><<<gA, 32, 0, stream>>>(C, sigma, HbA, AHb);
  k_convOut<<<dim3(B_ * N_ / 16, 256 / 16), 32, 0, stream>>>(
      AHb, WbT, bll[2], gl[2], bel[2], ml[2], vl[2], HbB, NDIL * 128, 256);

  k_pool<<<(B_ * (N_ / 4) * 256 + 255) / 256, 256, 0, stream>>>(HbB, P);
  k_zero<<<(8 * 512 + 255) / 256, 256, 0, stream>>>(FC1, 8 * 512);
  k_fc1<<<64, 512, 0, stream>>>(P, fcW1, FC1);
  k_fc2<<<1, 256, 0, stream>>>(FC1, fcb1, fcW2, fcb2, (float*)d_out);
}
